// MultiHeadAttention_61959198212542
// MI455X (gfx1250) — matrix operand order repaired and checked by reading, not compiled, not run on hardware
//
#include <hip/hip_runtime.h>
#include <hip/hip_bf16.h>

typedef _Float16 h8  __attribute__((ext_vector_type(8)));
typedef _Float16 h16 __attribute__((ext_vector_type(16)));
typedef float    f8  __attribute__((ext_vector_type(8)));

#define DMODEL 1024
#define NHEADS 16
#define HD     64
#define NB     16
#define TT     512
#define MROWS  (NB * TT)   // 8192

__device__ inline h16 cat8(h8 lo, h8 hi) {
    h16 r;
#pragma unroll
    for (int i = 0; i < 8; i++) { r[i] = lo[i]; r[i + 8] = hi[i]; }
    return r;
}

__device__ inline f8 zero8() {
    f8 r;
#pragma unroll
    for (int i = 0; i < 8; i++) r[i] = 0.0f;
    return r;
}

__device__ inline f8 wmma16(h16 a, h16 b, f8 c) {
    // D = A(16x32 f16) * B(32x16 f16) + C(16x16 f32)
    return __builtin_amdgcn_wmma_f32_16x16x32_f16(
        false, a, false, b, (short)0, c, false, false);
}

// async global->LDS 16-byte copy (gfx1250, ASYNCcnt-tracked)
__device__ inline void async_b128(const void* lds_dst, const void* gsrc) {
    unsigned lds_off = (unsigned)(size_t)lds_dst;   // flat LDS addr[31:0] = LDS offset
    unsigned long long ga = (unsigned long long)(size_t)gsrc;
    asm volatile("global_load_async_to_lds_b128 %0, %1, off"
                 :: "v"(lds_off), "v"(ga) : "memory");
}
__device__ inline void wait_async0() {
    asm volatile("s_wait_asynccnt 0x0" ::: "memory");
}

// ---------------------------------------------------------------------------
// fp32 -> f16 conversion (layout preserving)
// ---------------------------------------------------------------------------
__global__ void cvt_f32_f16(const float* __restrict__ src,
                            _Float16* __restrict__ dst, int n) {
    int i = blockIdx.x * blockDim.x + threadIdx.x;
    if (i < n) dst[i] = (_Float16)src[i];
}

// fp32 [K][N] -> f16 transposed [N][K]
__global__ void cvt_transpose_f16(const float* __restrict__ src,
                                  _Float16* __restrict__ dst, int K, int N) {
    int i = blockIdx.x * blockDim.x + threadIdx.x;
    if (i < K * N) {
        int k = i / N;
        int n = i - k * N;
        dst[(size_t)n * K + k] = (_Float16)src[i];
    }
}

// ---------------------------------------------------------------------------
// Tiled GEMM: C[M=8192, N] = A[M,1024] * BT^T + bias, BT is f16 [N][K]
// mode 0: scatter f16 into Q/K [B,H,T,hd] and V transposed [B,H,hd,T] (N=3072)
// mode 1: store f32 into outf [M,N]                                   (N=1024)
// Block: 256 threads (8 waves). Tile 128x128, double-buffered async LDS.
// ---------------------------------------------------------------------------
__global__ __launch_bounds__(256) void gemm_f16(
    const _Float16* __restrict__ A, const _Float16* __restrict__ BT,
    const float* __restrict__ bias, int N, int mode,
    _Float16* __restrict__ oq, _Float16* __restrict__ ok,
    _Float16* __restrict__ ov, float* __restrict__ outf) {
    __shared__ _Float16 sA[2][128 * 32];    // [row][k]
    __shared__ _Float16 sB[2][128 * 32];    // [n][k]

    const int K = DMODEL;
    const int nTilesN = N / 128;
    const int tn = blockIdx.x % nTilesN;
    const int tm = blockIdx.x / nTilesN;
    const int tid = threadIdx.x;
    const int lane = tid & 31;
    const int wid = tid >> 5;
    const int wm = wid >> 2;   // 0..1
    const int wn = wid & 3;    // 0..3
    const int g = lane >> 4;
    const int l15 = lane & 15;

    const int rowBase = tm * 128;
    const int colBase = tn * 128;

    f8 acc[4][2];
#pragma unroll
    for (int i = 0; i < 4; i++)
#pragma unroll
        for (int j = 0; j < 2; j++) acc[i][j] = zero8();

    // stage tile (128x32 halves each for A and B), 16B chunks, 2+2 per thread
    auto stage = [&](int ko, int buf) {
#pragma unroll
        for (int c = 2 * tid; c < 2 * tid + 2; c++) {
            int r = c >> 2;
            int p = (c & 3) * 8;
            async_b128(&sA[buf][r * 32 + p],
                       A + (size_t)(rowBase + r) * K + ko + p);
            async_b128(&sB[buf][r * 32 + p],
                       BT + (size_t)(colBase + r) * K + ko + p);
        }
    };

    const int nK = K / 32;
    int cur = 0;
    stage(0, 0);

    for (int i = 0; i < nK; i++) {
        wait_async0();
        __syncthreads();
        if (i + 1 < nK) stage((i + 1) * 32, cur ^ 1);

        const _Float16* lA = sA[cur];
        const _Float16* lB = sB[cur];
        h16 af[4];
#pragma unroll
        for (int mi = 0; mi < 4; mi++) {
            int r = wm * 64 + mi * 16 + l15;
            h8 lo = *(const h8*)(&lA[r * 32 + 8 * g]);
            h8 hi = *(const h8*)(&lA[r * 32 + 16 + 8 * g]);
            af[mi] = cat8(lo, hi);
        }
        h16 bf[2];
#pragma unroll
        for (int ni = 0; ni < 2; ni++) {
            int n = wn * 32 + ni * 16 + l15;
            h8 lo = *(const h8*)(&lB[n * 32 + 8 * g]);
            h8 hi = *(const h8*)(&lB[n * 32 + 16 + 8 * g]);
            bf[ni] = cat8(lo, hi);
        }
#pragma unroll
        for (int mi = 0; mi < 4; mi++)
#pragma unroll
            for (int ni = 0; ni < 2; ni++)
                acc[mi][ni] = wmma16(af[mi], bf[ni], acc[mi][ni]);
        __syncthreads();
        cur ^= 1;
    }

#pragma unroll
    for (int ni = 0; ni < 2; ni++) {
        int c = colBase + wn * 32 + ni * 16 + l15;
        float bv = bias[c];
        if (mode == 0) {
            int which = c / DMODEL;
            int cc = c - which * DMODEL;
            int hh = cc >> 6;
            int dd = cc & 63;
#pragma unroll
            for (int mi = 0; mi < 4; mi++) {
#pragma unroll
                for (int v = 0; v < 8; v++) {
                    int r = rowBase + wm * 64 + mi * 16 + v + 8 * g;
                    int bb = r >> 9;
                    int t = r & 511;
                    float val = acc[mi][ni][v] + bv;
                    _Float16 hv = (_Float16)val;
                    if (which == 0)
                        oq[(((size_t)(bb * NHEADS + hh) * TT + t) * HD + dd)] = hv;
                    else if (which == 1)
                        ok[(((size_t)(bb * NHEADS + hh) * TT + t) * HD + dd)] = hv;
                    else   // V transposed: [B,H,hd,T]
                        ov[(((size_t)(bb * NHEADS + hh) * HD + dd) * TT + t)] = hv;
                }
            }
        } else {
#pragma unroll
            for (int mi = 0; mi < 4; mi++) {
#pragma unroll
                for (int v = 0; v < 8; v++) {
                    int r = rowBase + wm * 64 + mi * 16 + v + 8 * g;
                    outf[(size_t)r * N + c] = acc[mi][ni][v] + bv;
                }
            }
        }
    }
}

// ---------------------------------------------------------------------------
// Flash-style causal attention. 1 block per (b, h, 64-row q tile).
// 128 threads = 4 waves; wave w owns q rows [q0+16w, q0+16w+16).
// K [B,H,T,hd], V pre-transposed [B,H,hd,T]; both async-staged to LDS.
// ---------------------------------------------------------------------------
__global__ __launch_bounds__(128) void attn_kernel(
    const _Float16* __restrict__ Qg, const _Float16* __restrict__ Kg,
    const _Float16* __restrict__ Vtg, _Float16* __restrict__ Y) {
    __shared__ _Float16 lK[64 * 64];        // [kv][d]
    __shared__ _Float16 lVt[64 * 64];       // [d][kv]
    __shared__ _Float16 lP[4 * 16 * 64];    // [wave][row][kv]

    const int bx = blockIdx.x;
    const int qt = bx & 7;
    const int hh = (bx >> 3) & 15;
    const int bb = bx >> 7;
    const int tid = threadIdx.x;
    const int lane = tid & 31;
    const int w = tid >> 5;
    const int g = lane >> 4;
    const int l15 = lane & 15;

    const size_t base = ((size_t)(bb * NHEADS + hh)) * TT * HD;
    const _Float16* q = Qg + base;
    const _Float16* kp = Kg + base;
    const _Float16* vtp = Vtg + base;   // [hd][T]

    const int q0 = qt * 64;
    const int qw = q0 + w * 16;

    h16 qf[2];
#pragma unroll
    for (int kk = 0; kk < 2; kk++) {
        const _Float16* src = q + (size_t)(qw + l15) * HD + kk * 32;
        h8 lo = *(const h8*)(src + 8 * g);
        h8 hi = *(const h8*)(src + 16 + 8 * g);
        qf[kk] = cat8(lo, hi);
    }

    f8 O[4];
#pragma unroll
    for (int dn = 0; dn < 4; dn++) O[dn] = zero8();
    float mrow[8], lrow[8];
#pragma unroll
    for (int i = 0; i < 8; i++) { mrow[i] = -1e30f; lrow[i] = 0.0f; }

    const float sc = 0.03125f;   // 1/sqrt(d_model=1024)

    for (int j = 0; j <= qt; j++) {
        const int jb = j * 64;
        __syncthreads();   // previous tile fully consumed before overwrite
        // async stage: K tile [kv][d] (row-major copy), Vt tile [d][kv]
#pragma unroll
        for (int i = 0; i < 4; i++) {
            int c = tid + 128 * i;          // 512 chunks of 16B each
            {   // lK: row kv = c>>2, 16B part (c&3)
                int kv = c >> 2;
                int p = (c & 3) * 8;
                async_b128(&lK[kv * 64 + p],
                           kp + (size_t)(jb + kv) * HD + p);
            }
            {   // lVt: row d = c>>3, 16B part (c&7); global row stride TT
                int d = c >> 3;
                int p = (c & 7) * 8;
                async_b128(&lVt[d * 64 + p],
                           vtp + (size_t)d * TT + jb + p);
            }
        }
        wait_async0();
        __syncthreads();

        // S = Q * K^T  (lK rows are exactly the B fragments: Bt[n][k]=K[n][k])
        f8 s[4];
#pragma unroll
        for (int n = 0; n < 4; n++) {
            f8 accS = zero8();
#pragma unroll
            for (int kk = 0; kk < 2; kk++) {
                const _Float16* src =
                    &lK[(n * 16 + l15) * 64 + kk * 32 + 8 * g];
                h8 lo = *(const h8*)(src);
                h8 hi = *(const h8*)(src + 16);
                accS = wmma16(qf[kk], cat8(lo, hi), accS);
            }
            s[n] = accS;
        }

        // scale + causal mask (C layout: row = v + 8g, col = n*16 + l15)
#pragma unroll
        for (int n = 0; n < 4; n++) {
            int col = jb + n * 16 + l15;
#pragma unroll
            for (int vv = 0; vv < 8; vv++) {
                int row = qw + vv + 8 * g;
                float val = s[n][vv] * sc;
                s[n][vv] = (col <= row) ? val : -1e30f;
            }
        }

        // online softmax update (reduce across 16-lane halves)
        float pscale[8];
#pragma unroll
        for (int vv = 0; vv < 8; vv++) {
            float t = fmaxf(fmaxf(s[0][vv], s[1][vv]),
                            fmaxf(s[2][vv], s[3][vv]));
#pragma unroll
            for (int mb = 1; mb < 16; mb <<= 1)
                t = fmaxf(t, __shfl_xor(t, mb, 32));
            float mnew = fmaxf(mrow[vv], t);
            pscale[vv] = __expf(mrow[vv] - mnew);
            mrow[vv] = mnew;
            float rs = 0.0f;
#pragma unroll
            for (int n = 0; n < 4; n++) {
                float p = __expf(s[n][vv] - mnew);
                s[n][vv] = p;
                rs += p;
            }
#pragma unroll
            for (int mb = 1; mb < 16; mb <<= 1)
                rs += __shfl_xor(rs, mb, 32);
            lrow[vv] = lrow[vv] * pscale[vv] + rs;
        }
#pragma unroll
        for (int dn = 0; dn < 4; dn++)
#pragma unroll
            for (int vv = 0; vv < 8; vv++) O[dn][vv] *= pscale[vv];

        // P (C layout) -> LDS [row][kv]  (per-wave region, no barrier needed)
        _Float16* pw = &lP[w * 16 * 64];
#pragma unroll
        for (int n = 0; n < 4; n++) {
            int col = n * 16 + l15;
#pragma unroll
            for (int vv = 0; vv < 8; vv++)
                pw[(vv + 8 * g) * 64 + col] = (_Float16)s[n][vv];
        }

        // O += P * V
#pragma unroll
        for (int n2 = 0; n2 < 2; n2++) {
            h8 lo = *(const h8*)(&pw[l15 * 64 + n2 * 32 + 8 * g]);
            h8 hi = *(const h8*)(&pw[l15 * 64 + n2 * 32 + 16 + 8 * g]);
            h16 pf = cat8(lo, hi);
#pragma unroll
            for (int dn = 0; dn < 4; dn++) {
                h8 blo = *(const h8*)(&lVt[(dn * 16 + l15) * 64 + n2 * 32 + 8 * g]);
                h8 bhi = *(const h8*)(&lVt[(dn * 16 + l15) * 64 + n2 * 32 + 16 + 8 * g]);
                O[dn] = wmma16(pf, cat8(blo, bhi), O[dn]);
            }
        }
    }

    // write y[b, t, h*64 + d] as f16 for the output projection
#pragma unroll
    for (int dn = 0; dn < 4; dn++) {
        int col = hh * HD + dn * 16 + l15;
#pragma unroll
        for (int vv = 0; vv < 8; vv++) {
            int row = qw + vv + 8 * g;
            float val = O[dn][vv] / lrow[vv];
            Y[((size_t)(bb * TT + row)) * DMODEL + col] = (_Float16)val;
        }
    }
}

// ---------------------------------------------------------------------------
// Host-side launch
// ---------------------------------------------------------------------------
extern "C" void kernel_launch(void* const* d_in, const int* in_sizes, int n_in,
                              void* d_out, int out_size, void* d_ws,
                              size_t ws_size, hipStream_t stream) {
    const float* x     = (const float*)d_in[0];
    const float* w_qkv = (const float*)d_in[1];
    const float* b_qkv = (const float*)d_in[2];
    const float* w_o   = (const float*)d_in[3];
    const float* b_o   = (const float*)d_in[4];
    float* out = (float*)d_out;

    char* ws = (char*)d_ws;
    const size_t XOFF  = 0;                                   // x f16
    const size_t WQOFF = XOFF + (size_t)MROWS * DMODEL * 2;   // w_qkv^T f16
    const size_t WOOFF = WQOFF + (size_t)DMODEL * 3 * DMODEL * 2;  // w_o^T f16
    const size_t QOFF  = WOOFF + (size_t)DMODEL * DMODEL * 2;
    const size_t KOFF  = QOFF + (size_t)MROWS * DMODEL * 2;
    const size_t VOFF  = KOFF + (size_t)MROWS * DMODEL * 2;
    const size_t YOFF  = VOFF + (size_t)MROWS * DMODEL * 2;

    _Float16* xh   = (_Float16*)(ws + XOFF);
    _Float16* wqT  = (_Float16*)(ws + WQOFF);
    _Float16* woT  = (_Float16*)(ws + WOOFF);
    _Float16* qh   = (_Float16*)(ws + QOFF);
    _Float16* kh   = (_Float16*)(ws + KOFF);
    _Float16* vth  = (_Float16*)(ws + VOFF);
    _Float16* yh   = (_Float16*)(ws + YOFF);

    {
        int n = MROWS * DMODEL;
        cvt_f32_f16<<<(n + 255) / 256, 256, 0, stream>>>(x, xh, n);
    }
    {
        int n = DMODEL * 3 * DMODEL;
        cvt_transpose_f16<<<(n + 255) / 256, 256, 0, stream>>>(
            w_qkv, wqT, DMODEL, 3 * DMODEL);
    }
    {
        int n = DMODEL * DMODEL;
        cvt_transpose_f16<<<(n + 255) / 256, 256, 0, stream>>>(
            w_o, woT, DMODEL, DMODEL);
    }

    // QKV projection: 8192 x 3072, tiles 64 x 24
    gemm_f16<<<(MROWS / 128) * (3 * DMODEL / 128), 256, 0, stream>>>(
        xh, wqT, b_qkv, 3 * DMODEL, 0, qh, kh, vth, nullptr);

    // attention: B*H*(T/64) blocks
    attn_kernel<<<NB * NHEADS * (TT / 64), 128, 0, stream>>>(qh, kh, vth, yh);

    // output projection: 8192 x 1024, tiles 64 x 8
    gemm_f16<<<(MROWS / 128) * (DMODEL / 128), 256, 0, stream>>>(
        yh, woT, b_o, DMODEL, 1, nullptr, nullptr, nullptr, out);
}
